// GAT_76166950028624
// MI455X (gfx1250) — matrix operand order repaired and checked by reading, not compiled, not run on hardware
//
#include <hip/hip_runtime.h>

// ---------------- problem constants (match reference) ----------------
#define N_NODES 50000
#define N_EDGES 800000
#define E_TOT   850000          // edges + self loops
#define ED_DIM  16
#define HEADS   8
#define HDIM    32
#define HD      256             // HEADS*HDIM
#define N_MOL   500
#define SLOPE     0.2f
#define MLP_SLOPE 0.01f

typedef __attribute__((ext_vector_type(16))) _Float16 v16h;
typedef __attribute__((ext_vector_type(8)))  float    v8f;
typedef __attribute__((ext_vector_type(4)))  float    v4f;

// ---------------- helpers ----------------
__device__ __forceinline__ unsigned f2ord(float f) {
  unsigned u = __float_as_uint(f);
  return (u & 0x80000000u) ? ~u : (u | 0x80000000u);
}
__device__ __forceinline__ float ord2f(unsigned u) {
  return (u & 0x80000000u) ? __uint_as_float(u & 0x7fffffffu) : __uint_as_float(~u);
}

// ---------------- elementwise / setup kernels ----------------
__global__ void k_fill(float* p, float v, long n) {
  long i = (long)blockIdx.x * blockDim.x + threadIdx.x;
  if (i < n) p[i] = v;
}

// self-loop edge_attr = segment_sum of incoming edge_attr per node
__global__ void k_ealoop(const float* __restrict__ ea, const int* __restrict__ ei,
                         float* __restrict__ ealoop) {
  int t = blockIdx.x * blockDim.x + threadIdx.x;
  if (t >= N_EDGES * ED_DIM) return;
  int e = t >> 4, k = t & 15;
  int d = ei[N_EDGES + e];
  atomicAdd(&ealoop[d * ED_DIM + k], ea[t]);
}

// Bt[n*K + k] = (half) Wsrc[k*ldw + n]   (transpose + f32->f16)
__global__ void k_trconv(const float* __restrict__ Wsrc, int ldw,
                         _Float16* __restrict__ Bt, int K, int Nc) {
  int t = blockIdx.x * blockDim.x + threadIdx.x;
  if (t >= Nc * K) return;
  int n = t / K, k = t - n * K;
  Bt[t] = (_Float16)Wsrc[(size_t)k * ldw + n];
}

// ---------------- WMMA GEMM: C[M x Ncols] = A[M x K] @ B, f32 in/out ----------------
// Requirements (guaranteed by the launcher): M % 16 == 0, K % 32 == 0,
// Ncols == gridDim.y * 2 * NT * 16.  Bt is B pre-transposed to [Ncols x K] f16.
// 2 waves / block; each wave computes a 16 x (NT*16) strip: one A fragment is
// reused by NT back-to-back v_wmma (no D->A/B hazards between them).
template <int NT>
__global__ __launch_bounds__(64)
void k_gemm(const float* __restrict__ A, int lda,
            const _Float16* __restrict__ Bt,
            float* __restrict__ C, int ldc, int K) {
  const int lane = threadIdx.x & 31;
  const int wave = threadIdx.x >> 5;
  const int m    = blockIdx.x * 16 + (lane & 15);              // A row (always valid)
  const int nb0  = (blockIdx.y * 2 + wave) * NT * 16 + (lane & 15); // first column
  const int kA   = (lane >> 4) * 16;                           // A K-half select
  const int kB   = (lane >> 4) * 16;                           // B K-half select
  const float* arow = A + (size_t)m * lda;
  const _Float16* brow[NT];
#pragma unroll
  for (int t = 0; t < NT; ++t) brow[t] = Bt + (size_t)(nb0 + t * 16) * K;

  v8f acc[NT];
#pragma unroll
  for (int t = 0; t < NT; ++t) acc[t] = {};

  for (int k0 = 0; k0 < K; k0 += 32) {
    v16h a;
#pragma unroll
    for (int j = 0; j < 8; ++j) a[j]     = (_Float16)arow[k0 + kA + j];
#pragma unroll
    for (int j = 0; j < 8; ++j) a[j + 8] = (_Float16)arow[k0 + 8 + kA + j];
#pragma unroll
    for (int t = 0; t < NT; ++t) {
      v16h b = *reinterpret_cast<const v16h*>(brow[t] + k0 + kB); // 32B aligned
      acc[t] = __builtin_amdgcn_wmma_f32_16x16x32_f16(false, a, false, b,
                                                      (short)0, acc[t], false, false);
    }
  }
  const int rb = blockIdx.x * 16 + (lane >> 4) * 8;
#pragma unroll
  for (int t = 0; t < NT; ++t)
#pragma unroll
    for (int r = 0; r < 8; ++r)
      C[(size_t)(rb + r) * ldc + nb0 + t * 16] = acc[t][r];
}

// a_src[n,h], a_dst[n,h] from xl and attention vectors
__global__ void k_nodeatt(const float* __restrict__ xl, const float* __restrict__ as,
                          const float* __restrict__ ad, float* __restrict__ asrc,
                          float* __restrict__ adst) {
  int t = blockIdx.x * blockDim.x + threadIdx.x;
  if (t >= N_NODES * HEADS) return;
  int n = t >> 3, h = t & 7;
  const float* row = xl + (size_t)n * HD + h * HDIM;
  float s1 = 0.f, s2 = 0.f;
#pragma unroll
  for (int d2 = 0; d2 < HDIM; ++d2) {
    float v = row[d2];
    s1 += v * as[h * HDIM + d2];
    s2 += v * ad[h * HDIM + d2];
  }
  asrc[t] = s1; adst[t] = s2;
}

// M[k,h] = sum_d We[k, h*32+d] * ae[h,d]   (16x8, one tiny block)
__global__ void k_me(const float* __restrict__ We, const float* __restrict__ ae,
                     float* __restrict__ Me) {
  int t = threadIdx.x;
  if (t >= ED_DIM * HEADS) return;
  int k = t >> 3, h = t & 7;
  float s = 0.f;
#pragma unroll
  for (int d2 = 0; d2 < HDIM; ++d2) s += We[k * HD + h * HDIM + d2] * ae[h * HDIM + d2];
  Me[k * HEADS + h] = s;
}

// init layer output slice of Hcat with bias (aggregation atomically adds on top)
__global__ void k_biasinit(float* __restrict__ Hcat, int loff, const float* __restrict__ b) {
  int t = blockIdx.x * blockDim.x + threadIdx.x;
  if (t >= N_NODES * HD) return;
  int n = t >> 8, c = t & 255;
  Hcat[(size_t)n * 768 + loff + c] = b[c];
}

// pass A: alpha = leaky(a_src[src]+a_dst[dst]+ea@M), segment max via ordered-uint atomicMax
__global__ void k_alpha(const int* __restrict__ ei, const float* __restrict__ ea,
                        const float* __restrict__ ealoop, const float* __restrict__ Me,
                        const float* __restrict__ asrc, const float* __restrict__ adst,
                        float* __restrict__ alpha, unsigned* __restrict__ amax) {
  int t = blockIdx.x * blockDim.x + threadIdx.x;
  if (t >= E_TOT * HEADS) return;
  int e = t >> 3, h = t & 7;
  int s, d; const float* row;
  if (e < N_EDGES) { s = ei[e]; d = ei[N_EDGES + e]; row = ea + (size_t)e * ED_DIM; }
  else             { s = d = e - N_EDGES;            row = ealoop + (size_t)(e - N_EDGES) * ED_DIM; }
  float ag = 0.f;
#pragma unroll
  for (int k = 0; k < ED_DIM; ++k) ag += row[k] * Me[k * HEADS + h];
  float a = asrc[s * HEADS + h] + adst[d * HEADS + h] + ag;
  a = (a > 0.f) ? a : SLOPE * a;
  alpha[t] = a;
  atomicMax(&amax[d * HEADS + h], f2ord(a));
}

// pass B: ex = exp(alpha - max), segment sum
__global__ void k_exp(const int* __restrict__ ei, float* __restrict__ alpha,
                      const unsigned* __restrict__ amax, float* __restrict__ den) {
  int t = blockIdx.x * blockDim.x + threadIdx.x;
  if (t >= E_TOT * HEADS) return;
  int e = t >> 3, h = t & 7;
  int d = (e < N_EDGES) ? ei[N_EDGES + e] : (e - N_EDGES);
  float ex = expf(alpha[t] - ord2f(amax[d * HEADS + h]));
  alpha[t] = ex;
  atomicAdd(&den[d * HEADS + h], ex);
}

// pass C: out[dst] += xl[src] * (ex/den[dst]); float4 per thread, 64 threads/edge
__global__ void k_agg(const int* __restrict__ ei, const float* __restrict__ alpha,
                      const float* __restrict__ den, const float* __restrict__ xl,
                      float* __restrict__ Hcat, int loff) {
  long t = (long)blockIdx.x * blockDim.x + threadIdx.x;
  if (t >= (long)E_TOT * 64) return;
  int e = (int)(t >> 6);
  int c = (int)(t & 63) << 2;     // channel base (float4)
  int h = c >> 5;
  int s, d;
  if (e < N_EDGES) { s = ei[e]; d = ei[N_EDGES + e]; } else { s = d = e - N_EDGES; }
  float att = alpha[e * HEADS + h] / den[d * HEADS + h];
  v4f xs = *reinterpret_cast<const v4f*>(xl + (size_t)s * HD + c);
  float* outp = Hcat + (size_t)d * 768 + loff + c;
  atomicAdd(outp + 0, xs[0] * att);
  atomicAdd(outp + 1, xs[1] * att);
  atomicAdd(outp + 2, xs[2] * att);
  atomicAdd(outp + 3, xs[3] * att);
}

// graph pooling of h3
__global__ void k_pool(const float* __restrict__ Hcat, const int* __restrict__ batch,
                       float* __restrict__ hpool) {
  int t = blockIdx.x * blockDim.x + threadIdx.x;
  if (t >= N_NODES * HD) return;
  int n = t >> 8, c = t & 255;
  atomicAdd(&hpool[batch[n] * HD + c], Hcat[(size_t)n * 768 + 512 + c]);
}

// p1[g,:] = lb1 + hpool[g,:] @ lw1[768:1024,:]  (tiny: 500x96)
__global__ void k_p1(const float* __restrict__ hpool, const float* __restrict__ lw1,
                     const float* __restrict__ lb1, float* __restrict__ p1) {
  int t = blockIdx.x * blockDim.x + threadIdx.x;
  if (t >= N_MOL * 96) return;
  int g = t / 96, j = t - g * 96;
  float s = lb1[j];
  const float* hp = hpool + g * HD;
  for (int k = 0; k < HD; ++k) s += hp[k] * lw1[(768 + k) * 96 + j];
  p1[t] = s;
}

__global__ void k_m1ep(float* __restrict__ m1, const float* __restrict__ p1,
                       const int* __restrict__ batch) {
  int t = blockIdx.x * blockDim.x + threadIdx.x;
  if (t >= N_NODES * 96) return;
  int n = t / 96, j = t - n * 96;
  float v = m1[t] + p1[batch[n] * 96 + j];
  m1[t] = (v > 0.f) ? v : MLP_SLOPE * v;
}

__global__ void k_m2ep(float* __restrict__ m2, const float* __restrict__ lb2) {
  int t = blockIdx.x * blockDim.x + threadIdx.x;
  if (t >= N_NODES * 64) return;
  int j = t & 63;
  float v = m2[t] + lb2[j];
  m2[t] = (v > 0.f) ? v : MLP_SLOPE * v;
}

__global__ void k_final(const float* __restrict__ m2, const float* __restrict__ lw3,
                        const float* __restrict__ lb3, float* __restrict__ out) {
  int n = blockIdx.x * blockDim.x + threadIdx.x;
  if (n >= N_NODES) return;
  const float* row = m2 + (size_t)n * 64;
  float s = lb3[0];
#pragma unroll
  for (int j = 0; j < 64; ++j) s += row[j] * lw3[j];
  out[n] = 1.f / (1.f + expf(-s));
}

// ---------------- driver ----------------
static inline unsigned nb(long n) { return (unsigned)((n + 255) / 256); }

extern "C" void kernel_launch(void* const* d_in, const int* in_sizes, int n_in,
                              void* d_out, int out_size, void* d_ws, size_t ws_size,
                              hipStream_t stream) {
  (void)in_sizes; (void)n_in; (void)out_size; (void)ws_size;
  const float* x          = (const float*)d_in[0];
  const int*   edge_index = (const int*)d_in[1];
  const float* edge_attr  = (const float*)d_in[2];
  const int*   batch      = (const int*)d_in[3];
  const float* Wl[3]  = {(const float*)d_in[4],  (const float*)d_in[10], (const float*)d_in[16]};
  const float* ASl[3] = {(const float*)d_in[5],  (const float*)d_in[11], (const float*)d_in[17]};
  const float* ADl[3] = {(const float*)d_in[6],  (const float*)d_in[12], (const float*)d_in[18]};
  const float* WEl[3] = {(const float*)d_in[7],  (const float*)d_in[13], (const float*)d_in[19]};
  const float* AEl[3] = {(const float*)d_in[8],  (const float*)d_in[14], (const float*)d_in[20]};
  const float* Bl[3]  = {(const float*)d_in[9],  (const float*)d_in[15], (const float*)d_in[21]};
  const float* lw1 = (const float*)d_in[22]; const float* lb1 = (const float*)d_in[23];
  const float* lw2 = (const float*)d_in[24]; const float* lb2 = (const float*)d_in[25];
  const float* lw3 = (const float*)d_in[26]; const float* lb3 = (const float*)d_in[27];
  float* out = (float*)d_out;
  float* ws  = (float*)d_ws;

  // workspace layout (floats; every offset is a multiple of 64 floats = 256B)
  float*    ealoop = ws;                         // N*16      = 800,000
  float*    Hcat   = ws + 800000;                // N*768     = 38,400,000
  float*    xl     = ws + 39200000;              // N*256     = 12,800,000
  float*    asrc   = ws + 52000000;              // N*8
  float*    adst   = ws + 52400000;              // N*8
  unsigned* amax   = (unsigned*)(ws + 52800000); // N*8
  float*    den    = ws + 53200000;              // N*8
  float*    alpha  = ws + 53600000;              // E_TOT*8   = 6,800,000
  float*    hpool  = ws + 60400000;              // NM*256
  float*    p1     = ws + 60528000;              // NM*96
  float*    m1     = ws + 60576000;              // N*96      = 4,800,000
  float*    m2     = ws + 65376000;              // N*64      = 3,200,000
  float*    Me     = ws + 68576000;              // 128
  _Float16* Bt     = (_Float16*)(ws + 68576128); // up to 96*768 halves

  // self-loop edge attrs + pooled accumulator init
  k_fill<<<nb((long)N_NODES * ED_DIM), 256, 0, stream>>>(ealoop, 0.f, (long)N_NODES * ED_DIM);
  k_fill<<<nb((long)N_MOL * HD), 256, 0, stream>>>(hpool, 0.f, (long)N_MOL * HD);
  k_ealoop<<<nb((long)N_EDGES * ED_DIM), 256, 0, stream>>>(edge_attr, edge_index, ealoop);

  for (int l = 0; l < 3; ++l) {
    const int Kin = (l == 0) ? 64 : 256;
    const float* Ain = (l == 0) ? x : (Hcat + (l - 1) * 256);
    const int lda    = (l == 0) ? 64 : 768;

    // xl = Ain @ W_l   (WMMA f16 x f16 -> f32);  Ncols=256: NT=4, 2 waves, grid.y=2
    k_trconv<<<nb((long)HD * Kin), 256, 0, stream>>>(Wl[l], HD, Bt, Kin, HD);
    k_gemm<4><<<dim3(N_NODES / 16, 2), 64, 0, stream>>>(Ain, lda, Bt, xl, HD, Kin);

    // attention scalars
    k_nodeatt<<<nb((long)N_NODES * HEADS), 256, 0, stream>>>(xl, ASl[l], ADl[l], asrc, adst);
    k_fill<<<nb((long)N_NODES * HEADS), 256, 0, stream>>>((float*)amax, 0.f, (long)N_NODES * HEADS);
    k_fill<<<nb((long)N_NODES * HEADS), 256, 0, stream>>>(den, 0.f, (long)N_NODES * HEADS);
    k_biasinit<<<nb((long)N_NODES * HD), 256, 0, stream>>>(Hcat, l * 256, Bl[l]);
    k_me<<<1, 128, 0, stream>>>(WEl[l], AEl[l], Me);

    // segment softmax + weighted aggregation
    k_alpha<<<nb((long)E_TOT * HEADS), 256, 0, stream>>>(edge_index, edge_attr, ealoop, Me,
                                                         asrc, adst, alpha, amax);
    k_exp<<<nb((long)E_TOT * HEADS), 256, 0, stream>>>(edge_index, alpha, amax, den);
    k_agg<<<nb((long)E_TOT * 64), 256, 0, stream>>>(edge_index, alpha, den, xl, Hcat, l * 256);
  }

  // readout
  k_pool<<<nb((long)N_NODES * HD), 256, 0, stream>>>(Hcat, batch, hpool);
  k_p1<<<nb((long)N_MOL * 96), 256, 0, stream>>>(hpool, lw1, lb1, p1);

  // m1 = Hcat @ lw1[0:768,:]  (+ p1[batch], leaky);  Ncols=96: NT=3, grid.y=1
  k_trconv<<<nb((long)96 * 768), 256, 0, stream>>>(lw1, 96, Bt, 768, 96);
  k_gemm<3><<<dim3(N_NODES / 16, 1), 64, 0, stream>>>(Hcat, 768, Bt, m1, 96, 768);
  k_m1ep<<<nb((long)N_NODES * 96), 256, 0, stream>>>(m1, p1, batch);

  // m2 = leaky(m1 @ lw2 + lb2);  Ncols=64: NT=2, grid.y=1
  k_trconv<<<nb((long)64 * 96), 256, 0, stream>>>(lw2, 64, Bt, 96, 64);
  k_gemm<2><<<dim3(N_NODES / 16, 1), 64, 0, stream>>>(m1, 96, Bt, m2, 64, 96);
  k_m2ep<<<nb((long)N_NODES * 64), 256, 0, stream>>>(m2, lb2);

  // out = sigmoid(m2 @ lw3 + lb3)
  k_final<<<nb((long)N_NODES), 256, 0, stream>>>(m2, lw3, lb3, out);
}
